// IterativeSequential2D_64123861729349
// MI455X (gfx1250) — matrix operand order repaired and checked by reading, not compiled, not run on hardware
//
#include <hip/hip_runtime.h>
#include <hip/hip_bf16.h>
#include <cstdint>

// ---------------------------------------------------------------------------
// Block-sparse iterative layer for MI455X (gfx1250, wave32, WMMA).
//
// - State kept as f16 ping-pong buffers in workspace (state+weights fit in
//   the 192MB L2, so inner iterations never touch HBM).
// - GEMMs on v_wmma_f32_16x16x32_f16 with f32 accumulation.
// - W tiles staged into LDS once per workgroup via GLOBAL_LOAD_ASYNC_TO_LDS
//   (CDNA5 async path, ASYNCcnt + s_wait_asynccnt), double-buffered so the
//   next contributing block's tile prefetches during compute.
// - bias + segment-sum + ReLU fused into the GEMM epilogue.
// ---------------------------------------------------------------------------

typedef __attribute__((ext_vector_type(16))) _Float16 v16h;
typedef __attribute__((ext_vector_type(8)))  _Float16 v8h;
typedef __attribute__((ext_vector_type(8)))  float    v8f;
typedef __attribute__((ext_vector_type(4)))  float    v4f;
typedef __attribute__((ext_vector_type(4)))  _Float16 v4h;

#define NBLK 16
#define DD   256
#define BB   8192
#define PP   60
#define LDS_PITCH 264   // 256 + 8 halfs: row stride 528B = 132 dwords -> bank-conflict-free

// Static sparsity pattern: block (i,j) present iff j==0 or |i-j|<=1,
// enumerated i-major / j-ascending (matches reference _PAIRS order).
__device__ __constant__ int c_cnt[NBLK] = {2,3,4,4,4,4,4,4,4,4,4,4,4,4,4,3};
__device__ __constant__ int c_pidx[NBLK][4] = {
  { 0, 1, 0, 0},{ 2, 3, 4, 0},{ 5, 6, 7, 8},{ 9,10,11,12},
  {13,14,15,16},{17,18,19,20},{21,22,23,24},{25,26,27,28},
  {29,30,31,32},{33,34,35,36},{37,38,39,40},{41,42,43,44},
  {45,46,47,48},{49,50,51,52},{53,54,55,56},{57,58,59, 0}};
__device__ __constant__ int c_pcol[NBLK][4] = {
  {0, 1, 0, 0},{0, 1, 2, 0},{0, 1, 2, 3},{0, 2, 3, 4},
  {0, 3, 4, 5},{0, 4, 5, 6},{0, 5, 6, 7},{0, 6, 7, 8},
  {0, 7, 8, 9},{0, 8, 9,10},{0, 9,10,11},{0,10,11,12},
  {0,11,12,13},{0,12,13,14},{0,13,14,15},{0,14,15, 0}};

// CDNA5 async global->LDS copy (16B per lane), tracked by ASYNCcnt.
// Builtins don't cover this; inline asm per CDNA5 ISA.
__device__ __forceinline__ void async_copy_b128(uint32_t lds_byte_addr,
                                                const _Float16* gaddr) {
  asm volatile("global_load_async_to_lds_b128 %0, %1, off"
               :: "v"(lds_byte_addr), "v"((uint64_t)(uintptr_t)gaddr)
               : "memory");
}
__device__ __forceinline__ void wait_async_le8() {
  asm volatile("s_wait_asynccnt 8" ::: "memory");
}
__device__ __forceinline__ void wait_async_0() {
  asm volatile("s_wait_asynccnt 0" ::: "memory");
}

// ---------------------------------------------------------------------------
// f32 -> f16 conversion (vectorized x4), used once per call for W and X.
// ---------------------------------------------------------------------------
__global__ void cvt_f32_to_f16_x4(const v4f* __restrict__ src,
                                  v4h* __restrict__ dst, int n4) {
  int i = blockIdx.x * blockDim.x + threadIdx.x;
  if (i < n4) {
    v4f x = src[i];
    v4h h;
    h[0] = (_Float16)x[0]; h[1] = (_Float16)x[1];
    h[2] = (_Float16)x[2]; h[3] = (_Float16)x[3];
    dst[i] = h;
  }
}

// ---------------------------------------------------------------------------
// One iteration: for output block `blk`, accumulate all contributing block
// GEMMs (Y = S[col] @ W[p]^T) + biases in f32 WMMA accumulators, ReLU, store.
//
// Wave tile: 32 (batch) x 64 (out features) = 2x4 accumulators of 16x16.
// Workgroup: 8 waves stacked along batch, shared e0 -> shared W tile in LDS.
// Grid: (BB/256, DD/64, NBLK).
// ---------------------------------------------------------------------------
template<bool FIRST, bool LAST>
__global__ void __launch_bounds__(256)
bs_iter_kernel(const _Float16* __restrict__ Ain,   // FIRST: [B,D] else [16,B,D] f16
               const _Float16* __restrict__ Wh,    // [P,D,D] f16 (out-major rows)
               const float*    __restrict__ bias,  // [P,D]   f32
               _Float16*       __restrict__ Sout,  // [16,B,D] f16 (if !LAST)
               float*          __restrict__ Fout)  // [16,B,D] f32 (if LAST)
{
  __shared__ _Float16 wtile[2][64][LDS_PITCH];   // double-buffered 64x256 W tile

  const int lane = threadIdx.x & 31;
  const int wave = threadIdx.x >> 5;
  const int lrow = lane & 15;   // m (A) / n (B,C,D) within 16
  const int lhi  = lane >> 4;   // lane-half selector for K split / M split

  const int blk    = blockIdx.z;
  const int e0     = blockIdx.y * 64;
  const int batch0 = blockIdx.x * 256 + wave * 32;

  const int ncontrib = FIRST ? 1 : c_cnt[blk];

  // Bias: segment-sum of biases of all contributing blocks, per out feature.
  float bsum[4];
#pragma unroll
  for (int nb = 0; nb < 4; ++nb) bsum[nb] = 0.0f;
  for (int c = 0; c < ncontrib; ++c) {
    const int p = c_pidx[blk][c];
#pragma unroll
    for (int nb = 0; nb < 4; ++nb)
      bsum[nb] += bias[p * DD + e0 + nb * 16 + lrow];
  }

  v8f acc[2][4];
#pragma unroll
  for (int mb = 0; mb < 2; ++mb)
#pragma unroll
    for (int nb = 0; nb < 4; ++nb)
#pragma unroll
      for (int r = 0; r < 8; ++r)
        acc[mb][nb][r] = bsum[nb];

  const int klo = lhi * 8;   // A-operand K sub-offset per ISA 16-bit A layout

  // Stage W tile for pair 0: wave w, async-instr q -> row q*8+w, 16B per lane.
  {
    const _Float16* Wp = Wh + (size_t)c_pidx[blk][0] * DD * DD;
#pragma unroll
    for (int q = 0; q < 8; ++q) {
      const int row = q * 8 + wave;
      async_copy_b128((uint32_t)(uintptr_t)&wtile[0][row][lane * 8],
                      Wp + (size_t)(e0 + row) * DD + lane * 8);
    }
  }

  for (int c = 0; c < ncontrib; ++c) {
    const int j = c_pcol[blk][c];
    const _Float16* __restrict__ A =
        FIRST ? Ain : (Ain + (size_t)j * BB * DD);
    const _Float16* arow0 = A + (size_t)(batch0 + lrow) * DD;       // mb=0
    const _Float16* arow1 = arow0 + (size_t)16 * DD;                // mb=1

    // Close previous compute phase: buf[(c+1)&1] free to overwrite.
    __syncthreads();
    if (c + 1 < ncontrib) {
      const _Float16* Wp = Wh + (size_t)c_pidx[blk][c + 1] * DD * DD;
#pragma unroll
      for (int q = 0; q < 8; ++q) {
        const int row = q * 8 + wave;
        async_copy_b128((uint32_t)(uintptr_t)&wtile[(c + 1) & 1][row][lane * 8],
                        Wp + (size_t)(e0 + row) * DD + lane * 8);
      }
      wait_async_le8();   // async loads retire in order: tile c is resident
    } else {
      wait_async_0();
    }
    __syncthreads();      // publish all waves' slices of buf[c&1]

    // B-operand rows from LDS: lane n = lane%16; K = 8h..8h+7, 8h+16..8h+23.
    const _Float16* brow[4];
#pragma unroll
    for (int nb = 0; nb < 4; ++nb)
      brow[nb] = &wtile[c & 1][nb * 16 + lrow][lhi * 8];

    union U { v16h v; v8h h[2]; };

#pragma unroll
    for (int k0 = 0; k0 < DD; k0 += 32) {
      // A tiles (16x32 f16): lane m = lane%16; K = klo..klo+7, klo+16..klo+23
      U ua0, ua1;
      ua0.h[0] = *(const v8h*)(arow0 + k0 + klo);
      ua0.h[1] = *(const v8h*)(arow0 + k0 + klo + 16);
      ua1.h[0] = *(const v8h*)(arow1 + k0 + klo);
      ua1.h[1] = *(const v8h*)(arow1 + k0 + klo + 16);
      v16h a[2] = { ua0.v, ua1.v };

      v16h bm[4];
#pragma unroll
      for (int nb = 0; nb < 4; ++nb) {
        U ub;
        ub.h[0] = *(const v8h*)(brow[nb] + k0);
        ub.h[1] = *(const v8h*)(brow[nb] + k0 + 16);
        bm[nb] = ub.v;
      }

#pragma unroll
      for (int mb = 0; mb < 2; ++mb)
#pragma unroll
        for (int nb = 0; nb < 4; ++nb)
          acc[mb][nb] = __builtin_amdgcn_wmma_f32_16x16x32_f16(
              /*neg_a=*/false, a[mb], /*neg_b=*/false, bm[nb],
              /*c_mod=*/(short)0, acc[mb][nb],
              /*reuse_a=*/false, /*reuse_b=*/false);
    }
  }

  // Epilogue: ReLU + store. C/D layout: VGPR r holds row m = r + 8*(lane/16).
  const size_t outbase = (size_t)blk * BB * DD;
#pragma unroll
  for (int mb = 0; mb < 2; ++mb) {
#pragma unroll
    for (int r = 0; r < 8; ++r) {
      const int m = batch0 + mb * 16 + lhi * 8 + r;
      const size_t rowoff = outbase + (size_t)m * DD + e0 + lrow;
#pragma unroll
      for (int nb = 0; nb < 4; ++nb) {
        float v = acc[mb][nb][r];
        v = v > 0.0f ? v : 0.0f;
        if (LAST) Fout[rowoff + nb * 16] = v;
        else      Sout[rowoff + nb * 16] = (_Float16)v;
      }
    }
  }
}

// ---------------------------------------------------------------------------
// Host-side orchestration. Inputs: d_in[0]=X [B,D] f32, d_in[1]=W [P,D,D] f32,
// d_in[2]=b [P,D] f32. Output: [16,B,D] f32.
// Workspace layout (all 16B aligned):
//   Wh  f16 [P,D,D]   7,864,320 B
//   Xh  f16 [B,D]     4,194,304 B
//   SA  f16 [16,B,D] 67,108,864 B
//   SB  f16 [16,B,D] 67,108,864 B   (total ~146.3 MB)
// ---------------------------------------------------------------------------
extern "C" void kernel_launch(void* const* d_in, const int* in_sizes, int n_in,
                              void* d_out, int out_size, void* d_ws, size_t ws_size,
                              hipStream_t stream) {
  const float* X = (const float*)d_in[0];
  const float* W = (const float*)d_in[1];
  const float* b = (const float*)d_in[2];
  float* out = (float*)d_out;

  char* ws = (char*)d_ws;
  const size_t WH_BYTES = (size_t)PP * DD * DD * sizeof(_Float16);
  const size_t XH_BYTES = (size_t)BB * DD * sizeof(_Float16);
  const size_t S_BYTES  = (size_t)NBLK * BB * DD * sizeof(_Float16);
  _Float16* Wh = (_Float16*)(ws);
  _Float16* Xh = (_Float16*)(ws + WH_BYTES);
  _Float16* SA = (_Float16*)(ws + WH_BYTES + XH_BYTES);
  _Float16* SB = (_Float16*)(ws + WH_BYTES + XH_BYTES + S_BYTES);

  // Convert W and X to f16 once.
  {
    const int nW4 = PP * DD * DD / 4;
    const int nX4 = BB * DD / 4;
    cvt_f32_to_f16_x4<<<(nW4 + 255) / 256, 256, 0, stream>>>(
        (const v4f*)W, (v4h*)Wh, nW4);
    cvt_f32_to_f16_x4<<<(nX4 + 255) / 256, 256, 0, stream>>>(
        (const v4f*)X, (v4h*)Xh, nX4);
  }

  const dim3 grid(BB / 256, DD / 64, NBLK);
  const dim3 blockDim(256);

  // Iteration 0: only (i,0) blocks, input X.
  bs_iter_kernel<true, false><<<grid, blockDim, 0, stream>>>(Xh, Wh, b, SA, nullptr);
  // Iterations 1..4: full pattern, f16 ping-pong state.
  bs_iter_kernel<false, false><<<grid, blockDim, 0, stream>>>(SA, Wh, b, SB, nullptr);
  bs_iter_kernel<false, false><<<grid, blockDim, 0, stream>>>(SB, Wh, b, SA, nullptr);
  bs_iter_kernel<false, false><<<grid, blockDim, 0, stream>>>(SA, Wh, b, SB, nullptr);
  bs_iter_kernel<false, false><<<grid, blockDim, 0, stream>>>(SB, Wh, b, SA, nullptr);
  // Iteration 5: write f32 result to d_out.
  bs_iter_kernel<false, true><<<grid, blockDim, 0, stream>>>(SA, Wh, b, nullptr, out);
}
